// GridSampleCrossBEVAttention_79654463471841
// MI455X (gfx1250) — compile-verified, matrix operand order checked by reading, not run on hardware
//
#include <hip/hip_runtime.h>
#include <hip/hip_bf16.h>

// ---------------------------------------------------------------------------
// Fused GridSampleCrossBEVAttention for gfx1250 (MI455X), round 4.
//   - QB=4 queries per WG: implicit-GEMM conv M=256 x N=128 x K=576,
//     v_wmma_f32_16x16x32_bf16, f32 accumulation.
//   - sched_group_barrier pins the k-step pipeline to
//       4x VMEM-read (A) -> 16x DS-read (B) -> 16x WMMA
//     so no VALU/address-calc lands in a WMMA's 4-slot WAR shadow
//     (the v_nop bubbles seen in rounds 2-3).
//   - LDS ~158 KB -> 2 workgroups resident per 320 KB WGP.
// ---------------------------------------------------------------------------

typedef __attribute__((ext_vector_type(16))) __bf16 v16bf;
typedef __attribute__((ext_vector_type(8)))  float  v8f;

#define HWD   128      // H == W
#define CIN   64
#define CMID  256      // conv output channels
#define TPTS  8
#define KQ    512
#define DQ    256
#define KDIM  576      // CIN * 9
#define QB    4        // queries per workgroup
#define NCORN (QB * TPTS * 4)   // 128 bilinear corners per WG
#define BROW  37       // padded corner row: 37*16 = 592 bf16 (576 + 16 pad)

// Scheduler pipeline hints (no-ops if the toolchain lacks the builtins).
#if __has_builtin(__builtin_amdgcn_sched_group_barrier)
#define PIPELINE_KSTEP()                                        \
  do {                                                          \
    __builtin_amdgcn_sched_group_barrier(0x020, 4, 0);  /* VMEM rd: A  */ \
    __builtin_amdgcn_sched_group_barrier(0x100, 16, 0); /* DS rd: B    */ \
    __builtin_amdgcn_sched_group_barrier(0x008, 16, 0); /* 16x WMMA    */ \
  } while (0)
#elif __has_builtin(__builtin_amdgcn_sched_barrier)
#define PIPELINE_KSTEP() __builtin_amdgcn_sched_barrier(0)
#else
#define PIPELINE_KSTEP() do {} while (0)
#endif

__device__ __forceinline__ unsigned short f2bf(float f) {
  union { float f; unsigned u; } x; x.f = f;
  unsigned r = (x.u + 0x7FFFu + ((x.u >> 16) & 1u)) >> 16;  // RNE
  return (unsigned short)r;
}

// Pre-pass: convert conv_w (256 x 576 fp32, [o][ci][ky][kx]) to bf16 in d_ws.
__global__ void convert_w_kernel(const float* __restrict__ w,
                                 unsigned short* __restrict__ o, int n) {
  int i = blockIdx.x * blockDim.x + threadIdx.x;
  if (i < n) o[i] = f2bf(w[i]);
}

__global__ __launch_bounds__(256) void bev_attn_kernel(
    const float* __restrict__ queries,       // (B,K,D)
    const float* __restrict__ traj,          // (B,K,T,2)
    const float* __restrict__ bev,           // (B,CIN,H,W)
    const unsigned short* __restrict__ wbf,  // (CMID,KDIM) bf16
    const float* __restrict__ conv_b,        // (CMID)
    const float* __restrict__ attn_W,        // (T,D)
    const float* __restrict__ attn_b,        // (T)
    const float* __restrict__ out_W,         // (D,D)
    const float* __restrict__ out_b,         // (D)
    float* __restrict__ out)                 // (B,K,D)
{
  __shared__ v16bf BmatV[NCORN][BROW];   // bf16 patch matrix  (151.5 KB)
  __shared__ float accC[QB][CMID];       // per-query channel accumulators
  __shared__ float convbL[CMID];
  __shared__ float logitsL[QB][TPTS];
  __shared__ float sW[NCORN];
  __shared__ int   cxL[NCORN], cyL[NCORN];

  const int kq0  = blockIdx.x * QB;
  const int b    = blockIdx.y;
  const int tid  = threadIdx.x;
  const int lane = tid & 31;
  const int wv   = tid >> 5;             // 8 waves
  const size_t qbase = ((size_t)b * KQ + kq0) * DQ;   // first query offset

  // ---- stage 1: bias to LDS, zero accumulators ----------------------------
  convbL[tid] = conv_b[tid];
#pragma unroll
  for (int qi = 0; qi < QB; ++qi) accC[qi][tid] = 0.f;
  __syncthreads();

  // ---- stage 2: attention logits (wave wv -> t = wv, loop over queries) ---
  {
    const float* wrow = attn_W + wv * DQ;
    float wl[8];
#pragma unroll
    for (int i = 0; i < 8; ++i) wl[i] = wrow[lane + 32 * i];
    const float bt = attn_b[wv];
#pragma unroll
    for (int qi = 0; qi < QB; ++qi) {
      const float* q = queries + qbase + (size_t)qi * DQ;
      float p = 0.f;
#pragma unroll
      for (int i = 0; i < 8; ++i) p += q[lane + 32 * i] * wl[i];
#pragma unroll
      for (int off = 16; off > 0; off >>= 1) p += __shfl_down(p, off);
      if (lane == 0) logitsL[qi][wv] = p + bt;
    }
  }
  __syncthreads();

  // ---- stage 3: softmax + bilinear corner weights (threads 0..127) --------
  if (tid < NCORN) {
    const int qi = tid >> 5;           // query within batch-of-4
    const int t  = (tid >> 2) & 7;     // sample point
    const int c  = tid & 3;            // bilinear corner
    float m = logitsL[qi][0];
#pragma unroll
    for (int i = 1; i < TPTS; ++i) m = fmaxf(m, logitsL[qi][i]);
    float den = 0.f;
#pragma unroll
    for (int i = 0; i < TPTS; ++i) den += __expf(logitsL[qi][i] - m);
    const float at = __expf(logitsL[qi][t] - m) / den;

    const float* tp = traj + (((size_t)b * KQ + kq0 + qi) * TPTS + t) * 2;
    const float gy = tp[0] * (1.f / 32.f);   // traj_x -> grid_y
    const float gx = tp[1] * (1.f / 32.f);   // traj_y -> grid_x
    const float x  = (gx + 1.f) * 64.f - 0.5f;   // align_corners=False
    const float y  = (gy + 1.f) * 64.f - 0.5f;
    const float x0 = floorf(x), y0 = floorf(y);
    const int dx = c & 1, dy = c >> 1;
    const float fx = x - x0, fy = y - y0;
    const float wx = dx ? fx : 1.f - fx;
    const float wy = dy ? fy : 1.f - fy;
    const int xi = (int)x0 + dx, yi = (int)y0 + dy;
    const bool valid = (unsigned)xi < HWD && (unsigned)yi < HWD;
    sW[tid]  = valid ? at * wx * wy : 0.f;   // fused attn * bilinear * pad
    cxL[tid] = xi;
    cyL[tid] = yi;
  }
  __syncthreads();

  // ---- stage 4: gather 3x3x64 patches -> bf16 B-matrix in LDS -------------
  {
    unsigned short* BmatU = (unsigned short*)BmatV;
    const float* bevB = bev + (size_t)b * CIN * HWD * HWD;
    for (int e = tid; e < NCORN * KDIM; e += 256) {
      const int j  = e / KDIM;
      const int k  = e - j * KDIM;
      const int ci = k / 9;
      const int r  = k - ci * 9;
      const int ky = r / 3;
      const int kx = r - ky * 3;
      const int yy = cyL[j] + ky - 1;      // SAME padding of the 3x3 conv
      const int xx = cxL[j] + kx - 1;
      float v = 0.f;
      if ((unsigned)yy < HWD && (unsigned)xx < HWD)
        v = bevB[(ci << 14) + (yy << 7) + xx];
      BmatU[j * (BROW * 16) + k] = f2bf(v);
    }
  }
  __syncthreads();

  // ---- stage 5: implicit-GEMM conv, 256x128x576, bf16 WMMA ----------------
  // wave wv owns M-tiles {wv, wv+8} x all 8 N-tiles: A fragments reused 8x.
  // sched_group_barrier enforces loads-then-WMMAs per k-step so the register
  // allocator keeps all 8 B fragments live (no WAR in the WMMA shadow).
  v8f acc[2][8];
#pragma unroll
  for (int mi = 0; mi < 2; ++mi)
#pragma unroll
    for (int n = 0; n < 8; ++n) acc[mi][n] = (v8f){};
  {
    const int half  = (lane >= 16) ? 16 : 0;   // K split across lane halves
    const int hv    = half >> 4;
    const int colL  = lane & 15;
    const unsigned short* aP0 = wbf + (size_t)(wv * 16 + colL) * KDIM;
    const unsigned short* aP1 = wbf + (size_t)((wv + 8) * 16 + colL) * KDIM;
#pragma unroll 1
    for (int ks = 0; ks < 18; ++ks) {
      const int kO   = ks * 32 + half;         // 32 B-aligned
      const int bIdx = ks * 2 + hv;
      v16bf a0 = *(const v16bf*)(aP0 + kO);
      v16bf a1 = *(const v16bf*)(aP1 + kO);
      v16bf bf[8];
#pragma unroll
      for (int n = 0; n < 8; ++n) bf[n] = BmatV[n * 16 + colL][bIdx];
#pragma unroll
      for (int n = 0; n < 8; ++n) {
        acc[0][n] = __builtin_amdgcn_wmma_f32_16x16x32_bf16(
            false, a0, false, bf[n], (short)0, acc[0][n], false, false);
        acc[1][n] = __builtin_amdgcn_wmma_f32_16x16x32_bf16(
            false, a1, false, bf[n], (short)0, acc[1][n], false, false);
      }
      PIPELINE_KSTEP();
    }
  }

  // ---- stage 6: bias + ReLU + fused weights, reduce over corners ----------
  {
    const int colL   = lane & 15;
    const int rowAdd = (lane >= 16) ? 8 : 0;   // C/D tile layout
#pragma unroll
    for (int n = 0; n < 8; ++n) {
      const int qi = n >> 1;                   // 2 N-tiles per query
      const float sj = sW[n * 16 + colL];
#pragma unroll
      for (int mi = 0; mi < 2; ++mi) {
        const int mbase = (wv + mi * 8) * 16;
#pragma unroll
        for (int r = 0; r < 8; ++r) {
          const int row = mbase + r + rowAdd;
          float v = fmaxf(acc[mi][n][r] + convbL[row], 0.f) * sj;
#pragma unroll
          for (int mk = 1; mk < 16; mk <<= 1) v += __shfl_xor(v, mk);
          if (colL == 0) atomicAdd(&accC[qi][row], v);   // ds_add_f32
        }
      }
    }
  }
  __syncthreads();

  // ---- stage 7: output projection + residual (out_W row reused 4x) --------
  {
    const float* wrow = out_W + (size_t)tid * DQ;   // out_W[d][c]
    float o[QB];
#pragma unroll
    for (int qi = 0; qi < QB; ++qi)
      o[qi] = out_b[tid] + queries[qbase + (size_t)qi * DQ + tid];
#pragma unroll 4
    for (int c = 0; c < DQ; ++c) {
      const float wc = wrow[c];
#pragma unroll
      for (int qi = 0; qi < QB; ++qi) o[qi] += wc * accC[qi][c];
    }
#pragma unroll
    for (int qi = 0; qi < QB; ++qi)
      out[qbase + (size_t)qi * DQ + tid] = o[qi];
  }
}

extern "C" void kernel_launch(void* const* d_in, const int* in_sizes, int n_in,
                              void* d_out, int out_size, void* d_ws, size_t ws_size,
                              hipStream_t stream) {
  const float* queries = (const float*)d_in[0];
  const float* traj    = (const float*)d_in[1];
  const float* bev     = (const float*)d_in[2];
  const float* conv_w  = (const float*)d_in[3];
  const float* conv_b  = (const float*)d_in[4];
  const float* attn_W  = (const float*)d_in[5];
  const float* attn_b  = (const float*)d_in[6];
  const float* out_W   = (const float*)d_in[7];
  const float* out_b   = (const float*)d_in[8];
  float* out = (float*)d_out;
  unsigned short* wbf = (unsigned short*)d_ws;   // 256*576*2 = 294,912 B

  const int nW = CMID * KDIM;
  convert_w_kernel<<<(nW + 255) / 256, 256, 0, stream>>>(conv_w, wbf, nW);

  dim3 grid(KQ / QB, 32);   // (k-group, b) -> 4096 workgroups
  bev_attn_kernel<<<grid, 256, 0, stream>>>(queries, traj, bev, wbf, conv_b,
                                            attn_W, attn_b, out_W, out_b, out);
}
